// MultiHeadAttention_73821897883749
// MI455X (gfx1250) — matrix operand order repaired and checked by reading, not compiled, not run on hardware
//
#include <hip/hip_runtime.h>

// ---------------------------------------------------------------------------
// MultiHeadAttention forward for MI455X (gfx1250, wave32, WMMA + TDM)
// B=4 T=2048 D=1024 H=16 Hd=64
// Pipeline: f32->f16 convert | TDM-staged QKV GEMM | flash attention | TDM-staged out GEMM
// ---------------------------------------------------------------------------

#define BATCH 4
#define SEQ   2048
#define DIM   1024
#define HEADS 16
#define HDIM  64
#define BT    (BATCH * SEQ)      // 8192
#define D3    (3 * DIM)          // 3072

typedef _Float16 half_t;
typedef unsigned int u32;
typedef unsigned long long u64;
typedef __attribute__((ext_vector_type(4)))  _Float16 v4h;
typedef __attribute__((ext_vector_type(8)))  _Float16 v8h;
typedef __attribute__((ext_vector_type(16))) _Float16 v16h;
typedef __attribute__((ext_vector_type(8)))  float    v8f;
typedef __attribute__((ext_vector_type(4)))  u32      u32x4;
typedef __attribute__((ext_vector_type(8)))  int      i32x8;
typedef __attribute__((ext_vector_type(4)))  int      i32x4;

static __device__ __forceinline__ v16h cat8(v8h lo, v8h hi) {
    v16h r;
#pragma unroll
    for (int i = 0; i < 8; ++i) { r[i] = lo[i]; r[8 + i] = hi[i]; }
    return r;
}

static __device__ __forceinline__ v8f wmma16(v16h a, v16h b, v8f c) {
    return __builtin_amdgcn_wmma_f32_16x16x32_f16(
        false, a, false, b, (short)0, c, false, false);
}

// ---------------------------------------------------------------------------
// TDM: 2D f16 tile (tile_d0 contiguous elems x tile_d1 rows) global -> LDS.
// LDS rows padded: pad_interval=16 DWORDs (code 3), pad_amount=4 DWORDs
// (code 3) -> LDS row stride 40 halves (80B): 16 consecutive rows map to
// distinct banks for the operand reads.  D# packed per ISA 8.3/8.4.
// ---------------------------------------------------------------------------
static __device__ __forceinline__ void tdm_load_2d(
    u32 lds_byte, u64 gaddr, u32 tensor_d0, u32 tensor_d1,
    u64 stride0_elems, u32 tile_d0, u32 tile_d1)
{
    u32x4 g0;
    g0[0] = 1u;                                               // count=1
    g0[1] = lds_byte;                                         // lds_addr
    g0[2] = (u32)(gaddr & 0xFFFFFFFFull);                     // global_addr[31:0]
    g0[3] = (u32)((gaddr >> 32) & 0x1FFFFFFull) | (2u << 30); // ga[56:32] | type=2
    i32x8 g1;
    g1[0] = (int)((1u << 16) | (1u << 20) | (3u << 22) | (3u << 25));
    g1[1] = (int)((tensor_d0 & 0xFFFFu) << 16);
    g1[2] = (int)(((tensor_d0 >> 16) & 0xFFFFu) | ((tensor_d1 & 0xFFFFu) << 16));
    g1[3] = (int)(((tensor_d1 >> 16) & 0xFFFFu) | ((tile_d0 & 0xFFFFu) << 16));
    g1[4] = (int)(tile_d1 & 0xFFFFu);
    g1[5] = (int)(u32)(stride0_elems & 0xFFFFFFFFull);
    g1[6] = (int)(u32)((stride0_elems >> 32) & 0xFFFFull);
    g1[7] = 0;
    i32x4 z4 = {};
#if defined(__clang_major__) && (__clang_major__ >= 23)
    i32x8 z8 = {};
    __builtin_amdgcn_tensor_load_to_lds(g0, g1, z4, z4, z8, 0);
#else
    __builtin_amdgcn_tensor_load_to_lds(g0, g1, z4, z4, 0);
#endif
}

static __device__ __forceinline__ u32 lds_off(const void* p) {
    return (u32)(uintptr_t)p;   // flat LDS address: addr[31:0] = LDS offset
}

#define LDSTRIDE 40   // halves per staged row (32 data + 4-DWORD pad)

// ---------------------------------------------------------------------------
// Kernel 0: f32 -> f16 conversion, 4 elements/thread, exact cover
// ---------------------------------------------------------------------------
__global__ __launch_bounds__(256) void mha_cvt_f16(
    const float* __restrict__ src, half_t* __restrict__ dst)
{
    const size_t i = ((size_t)blockIdx.x * 256 + threadIdx.x) * 4;
    float4 f = *(const float4*)(src + i);
    v4h h;
    h[0] = (half_t)f.x; h[1] = (half_t)f.y; h[2] = (half_t)f.z; h[3] = (half_t)f.w;
    *(v4h*)(dst + i) = h;
}

// ---------------------------------------------------------------------------
// Kernel 1: QKV GEMM.  C[BT,3D] = X16 * W16^T + bias -> scatter q/k[B,H,T,64],
// v[B,H,64,T] (transposed).  Block 128 thr = 128x64 tile; wave w -> 32 rows.
// ---------------------------------------------------------------------------
__global__ __launch_bounds__(128) void mha_qkv_gemm(
    const half_t* __restrict__ xh, const half_t* __restrict__ wh,
    const float* __restrict__ bias,
    half_t* __restrict__ qws, half_t* __restrict__ kws, half_t* __restrict__ vws)
{
    __shared__ half_t lA[2][128 * LDSTRIDE];
    __shared__ half_t lB[2][64 * LDSTRIDE];

    const int tid  = threadIdx.x;
    const int wave = tid >> 5;
    const int lane = tid & 31;
    const int col  = lane & 15;
    const int m0   = blockIdx.x * 128;
    const int n0   = blockIdx.y * 64;
    const bool issuer = (wave == 0);

    if (issuer) {
        tdm_load_2d(lds_off(&lA[0][0]), (u64)(uintptr_t)(xh + (size_t)m0 * DIM),
                    DIM, BT, DIM, 32, 128);
        tdm_load_2d(lds_off(&lB[0][0]), (u64)(uintptr_t)(wh + (size_t)n0 * DIM),
                    DIM, D3, DIM, 32, 64);
    }
    __builtin_amdgcn_s_wait_tensorcnt(0);
    __syncthreads();

    v8f acc[2][4] = {};
    const int kbA = (lane < 16) ? 0 : 8;
    const int kbB = (lane < 16) ? 0 : 8;

    for (int kt = 0; kt < DIM / 32; ++kt) {
        const int cur = kt & 1;
        if (kt + 1 < DIM / 32 && issuer) {
            const int k0 = (kt + 1) * 32;
            tdm_load_2d(lds_off(&lA[cur ^ 1][0]),
                        (u64)(uintptr_t)(xh + (size_t)m0 * DIM + k0),
                        DIM, BT, DIM, 32, 128);
            tdm_load_2d(lds_off(&lB[cur ^ 1][0]),
                        (u64)(uintptr_t)(wh + (size_t)n0 * DIM + k0),
                        DIM, D3, DIM, 32, 64);
        }
        const half_t* a0p = &lA[cur][(wave * 32 + col) * LDSTRIDE + kbA];
        const half_t* a1p = &lA[cur][(wave * 32 + 16 + col) * LDSTRIDE + kbA];
        v16h a0 = cat8(*(const v8h*)a0p, *(const v8h*)(a0p + 16));
        v16h a1 = cat8(*(const v8h*)a1p, *(const v8h*)(a1p + 16));
#pragma unroll
        for (int c4 = 0; c4 < 4; ++c4) {
            const half_t* bp = &lB[cur][(c4 * 16 + col) * LDSTRIDE + kbB];
            v16h b = cat8(*(const v8h*)bp, *(const v8h*)(bp + 16));
            acc[0][c4] = wmma16(a0, b, acc[0][c4]);
            acc[1][c4] = wmma16(a1, b, acc[1][c4]);
        }
        __builtin_amdgcn_s_wait_tensorcnt(0);
        __syncthreads();
    }

    const int rbase = (lane < 16) ? 0 : 8;
#pragma unroll
    for (int mc = 0; mc < 2; ++mc) {
#pragma unroll
        for (int c4 = 0; c4 < 4; ++c4) {
            const int n = n0 + c4 * 16 + col;
            const float bv = bias[n];
#pragma unroll
            for (int r = 0; r < 8; ++r) {
                const int row = m0 + wave * 32 + mc * 16 + rbase + r; // b*T + t
                const int b = row >> 11, t = row & (SEQ - 1);
                const half_t hv = (half_t)(acc[mc][c4][r] + bv);
                if (n < DIM) {
                    const int h = n >> 6, d = n & 63;
                    qws[((((size_t)b * HEADS + h) * SEQ) + t) * HDIM + d] = hv;
                } else if (n < 2 * DIM) {
                    const int nn = n - DIM, h = nn >> 6, d = nn & 63;
                    kws[((((size_t)b * HEADS + h) * SEQ) + t) * HDIM + d] = hv;
                } else {
                    const int nn = n - 2 * DIM, h = nn >> 6, d = nn & 63;
                    vws[((((size_t)b * HEADS + h) * HDIM) + d) * SEQ + t] = hv;
                }
            }
        }
    }
}

// ---------------------------------------------------------------------------
// Kernel 2: flash attention, one wave per 16-query tile, 32-key steps.
// ---------------------------------------------------------------------------
__global__ __launch_bounds__(32) void mha_attention(
    const half_t* __restrict__ qws, const half_t* __restrict__ kws,
    const half_t* __restrict__ vws, half_t* __restrict__ ows)
{
    __shared__ half_t pl[16 * 32];

    const int tile = blockIdx.x;
    const int qt   = tile & (SEQ / 16 - 1);
    const int bh   = tile / (SEQ / 16);
    const int lane = threadIdx.x;
    const int col  = lane & 15;
    const bool low = lane < 16;

    const half_t* Q = qws + (size_t)bh * SEQ * HDIM;
    const half_t* K = kws + (size_t)bh * SEQ * HDIM;
    const half_t* V = vws + (size_t)bh * HDIM * SEQ;

    const int kbA = low ? 0 : 8;
    const half_t* qp = Q + (size_t)(qt * 16 + col) * HDIM;
    v16h aq0 = cat8(*(const v8h*)(qp + kbA),      *(const v8h*)(qp + kbA + 16));
    v16h aq1 = cat8(*(const v8h*)(qp + 32 + kbA), *(const v8h*)(qp + 32 + kbA + 16));

    float m[8], l[8];
#pragma unroll
    for (int r = 0; r < 8; ++r) { m[r] = -1e30f; l[r] = 0.0f; }
    v8f o[4] = {};

    const int kbB = low ? 0 : 8;
    const int rbase = low ? 0 : 8;

    for (int j0 = 0; j0 < SEQ; j0 += 32) {
        const half_t* kp0 = K + (size_t)(j0 + col) * HDIM;
        const half_t* kp1 = K + (size_t)(j0 + 16 + col) * HDIM;
        v16h b00 = cat8(*(const v8h*)(kp0 + kbB),      *(const v8h*)(kp0 + kbB + 16));
        v16h b01 = cat8(*(const v8h*)(kp0 + 32 + kbB), *(const v8h*)(kp0 + 32 + kbB + 16));
        v16h b10 = cat8(*(const v8h*)(kp1 + kbB),      *(const v8h*)(kp1 + kbB + 16));
        v16h b11 = cat8(*(const v8h*)(kp1 + 32 + kbB), *(const v8h*)(kp1 + 32 + kbB + 16));
        v8f z = {};
        v8f s0 = wmma16(aq1, b01, wmma16(aq0, b00, z));
        v8f s1 = wmma16(aq1, b11, wmma16(aq0, b10, z));
#pragma unroll
        for (int r = 0; r < 8; ++r) { s0[r] *= 0.125f; s1[r] *= 0.125f; }

        float scl[8];
#pragma unroll
        for (int r = 0; r < 8; ++r) {
            float v = fmaxf(s0[r], s1[r]);
#pragma unroll
            for (int msk = 1; msk < 16; msk <<= 1) v = fmaxf(v, __shfl_xor(v, msk, 32));
            const float mn = fmaxf(m[r], v);
            scl[r] = __expf(m[r] - mn);
            m[r] = mn;
        }
#pragma unroll
        for (int r = 0; r < 8; ++r) {
            s0[r] = __expf(s0[r] - m[r]);
            s1[r] = __expf(s1[r] - m[r]);
            float rs = s0[r] + s1[r];
#pragma unroll
            for (int msk = 1; msk < 16; msk <<= 1) rs += __shfl_xor(rs, msk, 32);
            l[r] = l[r] * scl[r] + rs;
        }
#pragma unroll
        for (int c4 = 0; c4 < 4; ++c4)
#pragma unroll
            for (int r = 0; r < 8; ++r) o[c4][r] *= scl[r];

#pragma unroll
        for (int r = 0; r < 8; ++r) {
            pl[(rbase + r) * 32 + col]      = (half_t)s0[r];
            pl[(rbase + r) * 32 + 16 + col] = (half_t)s1[r];
        }
        __syncthreads();
        v16h pa = cat8(*(const v8h*)&pl[col * 32 + kbA],
                       *(const v8h*)&pl[col * 32 + kbA + 16]);

#pragma unroll
        for (int c4 = 0; c4 < 4; ++c4) {
            const half_t* vp = V + (size_t)(c4 * 16 + col) * SEQ + j0 + kbB;
            v16h vb = cat8(*(const v8h*)vp, *(const v8h*)(vp + 16));
            o[c4] = wmma16(pa, vb, o[c4]);
        }
        __syncthreads();
    }

    const int b = bh >> 4, h = bh & 15;
#pragma unroll
    for (int r = 0; r < 8; ++r) {
        const float inv = 1.0f / l[r];
        const int t = qt * 16 + rbase + r;
#pragma unroll
        for (int c4 = 0; c4 < 4; ++c4) {
            ows[((size_t)b * SEQ + t) * DIM + h * HDIM + c4 * 16 + col] =
                (half_t)(o[c4][r] * inv);
        }
    }
}

// ---------------------------------------------------------------------------
// Kernel 3: out projection.  Y[BT,D] = O16 * W16^T + out_b (fp32 out)
// ---------------------------------------------------------------------------
__global__ __launch_bounds__(128) void mha_out_gemm(
    const half_t* __restrict__ oin, const half_t* __restrict__ wh,
    const float* __restrict__ bias, float* __restrict__ out)
{
    __shared__ half_t lA[2][128 * LDSTRIDE];
    __shared__ half_t lB[2][64 * LDSTRIDE];

    const int tid  = threadIdx.x;
    const int wave = tid >> 5;
    const int lane = tid & 31;
    const int col  = lane & 15;
    const int m0   = blockIdx.x * 128;
    const int n0   = blockIdx.y * 64;
    const bool issuer = (wave == 0);

    if (issuer) {
        tdm_load_2d(lds_off(&lA[0][0]), (u64)(uintptr_t)(oin + (size_t)m0 * DIM),
                    DIM, BT, DIM, 32, 128);
        tdm_load_2d(lds_off(&lB[0][0]), (u64)(uintptr_t)(wh + (size_t)n0 * DIM),
                    DIM, DIM, DIM, 32, 64);
    }
    __builtin_amdgcn_s_wait_tensorcnt(0);
    __syncthreads();

    v8f acc[2][4] = {};
    const int kbA = (lane < 16) ? 0 : 8;
    const int kbB = (lane < 16) ? 0 : 8;

    for (int kt = 0; kt < DIM / 32; ++kt) {
        const int cur = kt & 1;
        if (kt + 1 < DIM / 32 && issuer) {
            const int k0 = (kt + 1) * 32;
            tdm_load_2d(lds_off(&lA[cur ^ 1][0]),
                        (u64)(uintptr_t)(oin + (size_t)m0 * DIM + k0),
                        DIM, BT, DIM, 32, 128);
            tdm_load_2d(lds_off(&lB[cur ^ 1][0]),
                        (u64)(uintptr_t)(wh + (size_t)n0 * DIM + k0),
                        DIM, DIM, DIM, 32, 64);
        }
        const half_t* a0p = &lA[cur][(wave * 32 + col) * LDSTRIDE + kbA];
        const half_t* a1p = &lA[cur][(wave * 32 + 16 + col) * LDSTRIDE + kbA];
        v16h a0 = cat8(*(const v8h*)a0p, *(const v8h*)(a0p + 16));
        v16h a1 = cat8(*(const v8h*)a1p, *(const v8h*)(a1p + 16));
#pragma unroll
        for (int c4 = 0; c4 < 4; ++c4) {
            const half_t* bp = &lB[cur][(c4 * 16 + col) * LDSTRIDE + kbB];
            v16h b = cat8(*(const v8h*)bp, *(const v8h*)(bp + 16));
            acc[0][c4] = wmma16(a0, b, acc[0][c4]);
            acc[1][c4] = wmma16(a1, b, acc[1][c4]);
        }
        __builtin_amdgcn_s_wait_tensorcnt(0);
        __syncthreads();
    }

    const int rbase = (lane < 16) ? 0 : 8;
#pragma unroll
    for (int mc = 0; mc < 2; ++mc) {
#pragma unroll
        for (int c4 = 0; c4 < 4; ++c4) {
            const int n = n0 + c4 * 16 + col;
            const float bv = bias[n];
#pragma unroll
            for (int r = 0; r < 8; ++r) {
                const int row = m0 + wave * 32 + mc * 16 + rbase + r;
                out[(size_t)row * DIM + n] = acc[mc][c4][r] + bv;
            }
        }
    }
}

// ---------------------------------------------------------------------------
extern "C" void kernel_launch(void* const* d_in, const int* in_sizes, int n_in,
                              void* d_out, int out_size, void* d_ws, size_t ws_size,
                              hipStream_t stream) {
    const float* x     = (const float*)d_in[0];
    const float* qkv_w = (const float*)d_in[1];
    const float* qkv_b = (const float*)d_in[2];
    const float* out_w = (const float*)d_in[3];
    const float* out_b = (const float*)d_in[4];
    float* out = (float*)d_out;

    // workspace (halves): x16 | qkv_w16 | out_w16 | q | k | v | o  (~92 MB)
    half_t* xh  = (half_t*)d_ws;
    half_t* w1h = xh  + (size_t)BT * DIM;
    half_t* w2h = w1h + (size_t)D3 * DIM;
    half_t* qws = w2h + (size_t)DIM * DIM;
    half_t* kws = qws + (size_t)BT * DIM;
    half_t* vws = kws + (size_t)BT * DIM;
    half_t* ows = vws + (size_t)BT * DIM;

    mha_cvt_f16<<<(BT * DIM) / 1024, 256, 0, stream>>>(x, xh);
    mha_cvt_f16<<<(D3 * DIM) / 1024, 256, 0, stream>>>(qkv_w, w1h);
    mha_cvt_f16<<<(DIM * DIM) / 1024, 256, 0, stream>>>(out_w, w2h);

    dim3 g1(BT / 128, D3 / 64);
    mha_qkv_gemm<<<g1, 128, 0, stream>>>(xh, w1h, qkv_b, qws, kws, vws);

    mha_attention<<<BATCH * HEADS * (SEQ / 16), 32, 0, stream>>>(qws, kws, vws, ows);

    dim3 g2(BT / 128, DIM / 64);
    mha_out_gemm<<<g2, 128, 0, stream>>>(ows, w2h, out_b, out);
}
